// AutoCorrelation_56573309223406
// MI455X (gfx1250) — matrix operand order repaired and checked by reading, not compiled, not run on hardware
//
#include <hip/hip_runtime.h>

// ---------------------------------------------------------------------------
// AutoCorrelation (B=8, L=S=H=D=E=128) for MI455X / gfx1250, wave32.
//   Phase 1: circular cross-correlation + softmax -> A (bf16) in workspace.
//   Phase 2: 1024x batched 128x128x128 GEMM via v_wmma_f32_16x16x32_bf16.
// Memory-bound (~320 MB of traffic ~ 14 us at 23.3 TB/s); bf16 intermediate
// halves workspace traffic and gives 8x K-depth/WMMA vs the f32 16x16x4 op.
// ---------------------------------------------------------------------------

typedef __attribute__((ext_vector_type(16))) __bf16          v16bf;
typedef __attribute__((ext_vector_type(8)))  __bf16          bf16x8;
typedef __attribute__((ext_vector_type(8)))  float           v8f;
typedef __attribute__((ext_vector_type(8)))  unsigned short  us8;

#define B_ 8
#define L_ 128
#define H_ 128
#define E_ 128

__device__ __forceinline__ unsigned short f32_to_bf16(float f) {
  union { float f; unsigned int u; } v; v.f = f;
  const unsigned int u = v.u;
  return (unsigned short)((u + 0x7FFFu + ((u >> 16) & 1u)) >> 16);  // RNE
}
__device__ __forceinline__ __bf16 bfbits(unsigned short u) {
  return __builtin_bit_cast(__bf16, u);
}

// ---------------------------------------------------------------------------
// Phase 1: cor[b,h,e,l] = sum_t q[b,t,h,e]*k[b,(t-l)&127,h,e]; softmax over l.
// One block per (b,h, e-block of 32). LDS tiles are transposed [e][t] so the
// global gathers are coalesced over e (lane dim) and the lag loop reads rows.
// ---------------------------------------------------------------------------
__global__ __launch_bounds__(256)
void ac_corr_softmax(const float* __restrict__ q,
                     const float* __restrict__ k,
                     unsigned short* __restrict__ Abf,   // bf16 bits [B][H][E][L]
                     float scale)
{
  constexpr int EB = 32;     // e-columns per block
  constexpr int TP = 132;    // padded t-stride (floats) -> bank-friendly
  __shared__ float Qs[EB][TP];
  __shared__ float Ks[EB][TP];

  const int blk  = blockIdx.x;
  const int eblk = blk & 3;           // E_/EB = 4
  const int bh   = blk >> 2;
  const int h    = bh & (H_ - 1);
  const int b    = bh >> 7;
  const int e0   = eblk * EB;
  const int tid  = (int)threadIdx.x;
  const int lane = tid & 31;
  const int wv   = tid >> 5;          // 8 waves

  // Cooperative coalesced load: lane sweeps contiguous e, waves sweep t.
  {
    const size_t colbase = (size_t)b * (L_ * H_ * E_) + (size_t)h * E_ + (size_t)(e0 + lane);
    const float* __restrict__ qp = q + colbase;
    const float* __restrict__ kp = k + colbase;
    for (int t = wv; t < L_; t += 8) {
      Qs[lane][t] = qp[(size_t)t * (H_ * E_)];
      Ks[lane][t] = kp[(size_t)t * (H_ * E_)];
    }
  }
  __syncthreads();

  // Each wave: 4 e-columns; each lane: 4 lags (l = 4*lane + j).
  for (int ec = 0; ec < 4; ++ec) {
    const int e = wv * 4 + ec;
    const float* __restrict__ Qr = Qs[e];
    const float* __restrict__ Kr = Ks[e];
    const int l0 = lane * 4;
    float a0 = 0.f, a1 = 0.f, a2 = 0.f, a3 = 0.f;
#pragma unroll 8
    for (int t = 0; t < L_; ++t) {
      const float qv = Qr[t];           // broadcast (same addr all lanes)
      const int base = t - l0;
      a0 = fmaf(qv, Kr[(base    ) & 127], a0);
      a1 = fmaf(qv, Kr[(base - 1) & 127], a1);
      a2 = fmaf(qv, Kr[(base - 2) & 127], a2);
      a3 = fmaf(qv, Kr[(base - 3) & 127], a3);
    }
    a0 *= scale; a1 *= scale; a2 *= scale; a3 *= scale;

    // Wave-wide softmax over the 128 lags (4 per lane).
    float m = fmaxf(fmaxf(a0, a1), fmaxf(a2, a3));
#pragma unroll
    for (int off = 16; off >= 1; off >>= 1)
      m = fmaxf(m, __shfl_xor(m, off, 32));
    const float p0 = __expf(a0 - m), p1 = __expf(a1 - m);
    const float p2 = __expf(a2 - m), p3 = __expf(a3 - m);
    float s = p0 + p1 + p2 + p3;
#pragma unroll
    for (int off = 16; off >= 1; off >>= 1)
      s += __shfl_xor(s, off, 32);
    const float inv = __builtin_amdgcn_rcpf(s);

    const unsigned int lo = (unsigned int)f32_to_bf16(p0 * inv) |
                            ((unsigned int)f32_to_bf16(p1 * inv) << 16);
    const unsigned int hi = (unsigned int)f32_to_bf16(p2 * inv) |
                            ((unsigned int)f32_to_bf16(p3 * inv) << 16);
    const size_t o = (((size_t)b * H_ + h) * E_ + (size_t)(e0 + e)) * (size_t)L_ + l0;
    uint2 pk; pk.x = lo; pk.y = hi;
    *(uint2*)(Abf + o) = pk;            // 8B packed, coalesced per half-wave
  }
}

// ---------------------------------------------------------------------------
// Phase 2: out[b,x,y,z] = sum_s A[b,x,z,s] * values[b,x,y,s]
// One block per (b,x): 128x128x128 GEMM. A slab staged bf16 in padded LDS
// (16B-aligned rows), V fragments pulled from global as float4 + cvt.
// Wave w owns output rows [16w,16w+16); A-fragment reused across 8 tz tiles.
// ---------------------------------------------------------------------------
__global__ __launch_bounds__(256)
void ac_av_wmma(const float* __restrict__ values,
                const unsigned short* __restrict__ Abf,
                float* __restrict__ out)
{
  constexpr int PS = 136;  // ushort stride: 272B rows = 16B aligned, 64-bank spread
  __shared__ unsigned short As[128][PS];

  const int bx = (int)blockIdx.x;                       // b*128 + x
  const float*          __restrict__ vsl = values + (size_t)bx * (H_ * E_);
  const unsigned short* __restrict__ asl = Abf    + (size_t)bx * (E_ * L_);
  float*                __restrict__ osl = out    + (size_t)bx * (H_ * E_);

  const int tid = (int)threadIdx.x;

  // Stage A slab (bf16, 32 KB) into LDS with 16B chunks, fully coalesced.
#pragma unroll
  for (int i = 0; i < 8; ++i) {
    const int g   = tid + i * 256;      // us8 chunk index (2048 total)
    const int row = g >> 4;             // 16 chunks per 128-elem row
    const int col = (g & 15) * 8;
    const us8 v = ((const us8*)asl)[g];
    *(us8*)&As[row][col] = v;
  }
  __syncthreads();

  const int lane = tid & 31;
  const int wv   = tid >> 5;
  const int m    = lane & 15;
  const int hif  = lane >> 4;           // half-wave selector
  const int ty   = wv;                  // wave's 16-row output strip

  v8f acc[8];
#pragma unroll
  for (int tz = 0; tz < 8; ++tz) { v8f z = {}; acc[tz] = z; }

  const float* __restrict__ vrow = vsl + (size_t)(ty * 16 + m) * E_;

#pragma unroll
  for (int k0 = 0; k0 < 128; k0 += 32) {
    // ---- A fragment: Vm[y=16*ty+m, k] in contiguous K order (matches B):
    // lanes 0-15 hold K=k0..k0+15; lanes 16-31 hold K=k0+16..k0+31.
    const int klo = k0 + hif * 16;
    const float4 f0 = *(const float4*)(vrow + klo);
    const float4 f1 = *(const float4*)(vrow + klo + 4);
    const float4 f2 = *(const float4*)(vrow + klo + 8);
    const float4 f3 = *(const float4*)(vrow + klo + 12);
    const float fv[16] = { f0.x, f0.y, f0.z, f0.w,  f1.x, f1.y, f1.z, f1.w,
                           f2.x, f2.y, f2.z, f2.w,  f3.x, f3.y, f3.z, f3.w };
    v16bf afrag;
#pragma unroll
    for (int i = 0; i < 16; ++i) afrag[i] = bfbits(f32_to_bf16(fv[i]));

    // ---- B fragments: B[k,n] = A[z=16*tz+n, s=k0+k]; lanes 0-15 K=0..15,
    // lanes 16-31 K=16..31 -> 16 consecutive s per lane = 2x ds_load_b128.
    const int kb = k0 + hif * 16;
#pragma unroll
    for (int tz = 0; tz < 8; ++tz) {
      const unsigned short* bp = &As[tz * 16 + m][kb];
      const bf16x8 blo = *(const bf16x8*)(bp);
      const bf16x8 bhi = *(const bf16x8*)(bp + 8);
      const v16bf bfrag = __builtin_shufflevector(
          blo, bhi, 0, 1, 2, 3, 4, 5, 6, 7, 8, 9, 10, 11, 12, 13, 14, 15);
      acc[tz] = __builtin_amdgcn_wmma_f32_16x16x32_bf16(
          false, afrag, false, bfrag, (short)0, acc[tz], false, false);
    }
  }

  // ---- Store D: VGPR r -> row (r + 8*half), col = lane%16.
#pragma unroll
  for (int tz = 0; tz < 8; ++tz) {
    const int z = tz * 16 + m;
#pragma unroll
    for (int r = 0; r < 8; ++r) {
      const int y = ty * 16 + r + hif * 8;
      osl[y * E_ + z] = acc[tz][r];
    }
  }
}

// ---------------------------------------------------------------------------
extern "C" void kernel_launch(void* const* d_in, const int* in_sizes, int n_in,
                              void* d_out, int out_size, void* d_ws, size_t ws_size,
                              hipStream_t stream) {
  (void)in_sizes; (void)n_in; (void)out_size; (void)ws_size;
  const float* q = (const float*)d_in[0];
  const float* k = (const float*)d_in[1];
  const float* v = (const float*)d_in[2];
  float* out = (float*)d_out;
  unsigned short* Abf = (unsigned short*)d_ws;   // needs B*H*E*L*2 = 32 MiB

  const float scale = 0.088388347648318447f;     // 1/sqrt(E=128)

  ac_corr_softmax<<<dim3(B_ * H_ * (E_ / 32)), dim3(256), 0, stream>>>(q, k, Abf, scale);
  ac_av_wmma<<<dim3(B_ * L_), dim3(256), 0, stream>>>(v, Abf, out);
}
